// MTNet_41721312313629
// MI455X (gfx1250) — compile-verified, matrix operand order checked by reading, not run on hardware
//
#include <hip/hip_runtime.h>
#include <hip/hip_bf16.h>

// ---------------------------------------------------------------------------
// MTNet memory-attention + prediction head for gfx1250 (CDNA5, wave32, WMMA)
// Shapes: N=16384 (B*L), H=256, M=4096, D=32, F=2H+D=544, TOPK=16
// ---------------------------------------------------------------------------

#define N_ROWS   16384
#define HDIM     256
#define MROWS    4096
#define DDIM     32
#define FDIM     544          // 2H + D
#define TOPK     16
#define MTILES   256          // MROWS / 16

typedef __attribute__((ext_vector_type(16))) __bf16 v16bf;
typedef __attribute__((ext_vector_type(8)))  float  v8f;

union BfPack { uint4 u[2]; v16bf v; };

// Load one 16x16x32 bf16 WMMA A/B fragment slice for this lane.
// Per-lane K set (ISA 7.12.2, 16-bit A 16x32 / B 32x16):
//   V0..V3 = k0 + {0..7},  V4..V7 = k0 + 16 + {0..7},  k0 = half*8 (+32*step)
// => two contiguous 16-byte loads at p and p+16 elements.
__device__ __forceinline__ v16bf load_bf16x16(const __hip_bfloat16* p) {
  BfPack b;
  b.u[0] = *reinterpret_cast<const uint4*>(p);
  b.u[1] = *reinterpret_cast<const uint4*>(p + 16);
  return b.v;
}

// ---------------------------------------------------------------------------
// Prep 1: row-wise L2 normalize u; write u_norm (f32 ws) and bf16 into feat
// cols [256..511]; also pack Q bf16 into feat cols [512..543].
// One block (256 threads) per row.
// ---------------------------------------------------------------------------
__global__ void __launch_bounds__(256) mtnet_prep_norm(
    const float* __restrict__ u, const float* __restrict__ Q,
    float* __restrict__ u_norm, __hip_bfloat16* __restrict__ feat) {
  const int row = blockIdx.x;
  const int t   = threadIdx.x;
  __shared__ float red[256];
  float x = u[(size_t)row * HDIM + t];
  red[t] = x * x;
  __syncthreads();
  #pragma unroll
  for (int s = 128; s > 0; s >>= 1) {
    if (t < s) red[t] += red[t + s];
    __syncthreads();
  }
  const float inv = 1.0f / fmaxf(sqrtf(red[0]), 1e-12f);
  const float un  = x * inv;
  u_norm[(size_t)row * HDIM + t] = un;
  feat[(size_t)row * FDIM + HDIM + t] = __float2bfloat16(un);
  if (t < DDIM)
    feat[(size_t)row * FDIM + 2 * HDIM + t] = __float2bfloat16(Q[(size_t)row * DDIM + t]);
}

// Prep 2: m -> bf16
__global__ void __launch_bounds__(256) mtnet_prep_mbf(
    const float* __restrict__ m, __hip_bfloat16* __restrict__ mb) {
  const int i = blockIdx.x * 256 + threadIdx.x;   // MROWS*HDIM / 256 blocks
  mb[i] = __float2bfloat16(m[i]);
}

// Prep 3: pred_w [F,H] -> wt [H,F] bf16 (transpose so B-fragment loads are rows)
__global__ void __launch_bounds__(256) mtnet_prep_wt(
    const float* __restrict__ w, __hip_bfloat16* __restrict__ wt) {
  const int k = blockIdx.x;    // 0..543
  const int n = threadIdx.x;   // 0..255
  wt[(size_t)n * FDIM + k] = __float2bfloat16(w[(size_t)k * HDIM + n]);
}

// ---------------------------------------------------------------------------
// Fused: scores (WMMA bf16) -> full 16x4096 stripe in LDS -> top-16 ->
// softmax -> o = sum p_k * m[idx_k] (f32) -> feat cols [0..255] as bf16.
// Grid: N/16 blocks, 256 threads (8 waves). Dynamic LDS ~290 KB (CDNA5 WGP
// allows up to 320 KB per workgroup).
// ---------------------------------------------------------------------------
#define SC_BYTES   (16 * MROWS * 4)          // 262144 : scores f32
#define CV_BYTES   (16 * 256 * 4)            //  16384 : candidate values
#define CI_BYTES   (16 * 256 * 4)            //  16384 : candidate indices
#define PB_BYTES   (16 * TOPK * 4)           //   1024 : softmax weights
#define IB_BYTES   (16 * TOPK * 4)           //   1024 : selected indices
#define K2_LDS     (SC_BYTES + CV_BYTES + CI_BYTES + PB_BYTES + IB_BYTES)

__global__ void __launch_bounds__(256) mtnet_score_topk_o(
    __hip_bfloat16* feat,                       // read u_norm cols, write o cols
    const __hip_bfloat16* __restrict__ mb,      // m in bf16 [MROWS, HDIM]
    const float* __restrict__ m) {              // m in f32  [MROWS, HDIM]
  extern __shared__ char smem[];
  float* scores = (float*)smem;
  float* candV  = (float*)(smem + SC_BYTES);
  int*   candI  = (int*)  (smem + SC_BYTES + CV_BYTES);
  float* pbuf   = (float*)(smem + SC_BYTES + CV_BYTES + CI_BYTES);
  int*   ibuf   = (int*)  (smem + SC_BYTES + CV_BYTES + CI_BYTES + PB_BYTES);

  const int t     = threadIdx.x;
  const int wave  = t >> 5;
  const int lane  = t & 31;
  const int half  = lane >> 4;
  const int l16   = lane & 15;
  const int grow0 = blockIdx.x * 16;           // first of 16 rows owned

  // ---- Phase 1: scores = u_norm @ m^T for this 16-row stripe ----
  // A fragments (same 16 rows for every M-tile) held in registers.
  v16bf afrag[8];
  {
    const __hip_bfloat16* ap =
        feat + (size_t)(grow0 + l16) * FDIM + HDIM + half * 8;
    #pragma unroll
    for (int kk = 0; kk < 8; ++kk) afrag[kk] = load_bf16x16(ap + kk * 32);
  }
  for (int j = wave; j < MTILES; j += 8) {     // 32 tiles per wave
    v8f acc = {};
    const __hip_bfloat16* bp = mb + (size_t)(j * 16 + l16) * HDIM + half * 8;
    #pragma unroll
    for (int kk = 0; kk < 8; ++kk) {
      v16bf bfrag = load_bf16x16(bp + kk * 32);
      acc = __builtin_amdgcn_wmma_f32_16x16x32_bf16(
          false, afrag[kk], false, bfrag, (short)0, acc, false, false);
    }
    const int col = j * 16 + l16;
    #pragma unroll
    for (int r = 0; r < 8; ++r)
      scores[(r + 8 * half) * MROWS + col] = acc[r];
  }
  __syncthreads();

  // ---- Phase 2: per-thread partial top-16 (16 threads per row) ----
  {
    const int row = t >> 4, sub = t & 15;
    float tv[TOPK]; int ti[TOPK];
    #pragma unroll
    for (int q = 0; q < TOPK; ++q) { tv[q] = -1e30f; ti[q] = 0; }
    const float* srow = scores + row * MROWS + sub * 256;
    for (int c = 0; c < 256; ++c) {
      const float v = srow[c];
      if (v > tv[TOPK - 1]) {
        int pos = TOPK - 1;
        while (pos > 0 && v > tv[pos - 1]) {
          tv[pos] = tv[pos - 1]; ti[pos] = ti[pos - 1]; --pos;
        }
        tv[pos] = v; ti[pos] = sub * 256 + c;
      }
    }
    #pragma unroll
    for (int q = 0; q < TOPK; ++q) {
      candV[row * 256 + sub * TOPK + q] = tv[q];
      candI[row * 256 + sub * TOPK + q] = ti[q];
    }
  }
  __syncthreads();

  // ---- Phase 3: merge 256 candidates/row, softmax over top-16 ----
  if (t < 16) {
    const int row = t;
    float tv[TOPK]; int ti[TOPK];
    #pragma unroll
    for (int q = 0; q < TOPK; ++q) { tv[q] = -1e30f; ti[q] = 0; }
    for (int c = 0; c < 256; ++c) {
      const float v = candV[row * 256 + c];
      if (v > tv[TOPK - 1]) {
        int pos = TOPK - 1;
        while (pos > 0 && v > tv[pos - 1]) {
          tv[pos] = tv[pos - 1]; ti[pos] = ti[pos - 1]; --pos;
        }
        tv[pos] = v; ti[pos] = candI[row * 256 + c];
      }
    }
    const float mx = tv[0];          // sorted descending
    float p[TOPK], s = 0.f;
    #pragma unroll
    for (int q = 0; q < TOPK; ++q) { p[q] = __expf(tv[q] - mx); s += p[q]; }
    const float inv = 1.0f / s;
    #pragma unroll
    for (int q = 0; q < TOPK; ++q) {
      pbuf[row * TOPK + q] = p[q] * inv;
      ibuf[row * TOPK + q] = ti[q];
    }
  }
  __syncthreads();

  // ---- Phase 4: o[row] = sum_k p_k * m[idx_k] (f32), store bf16 in feat ----
  {
    const int row = t >> 4, seg = t & 15;    // 16 H-columns per thread
    float oacc[16];
    #pragma unroll
    for (int hh = 0; hh < 16; ++hh) oacc[hh] = 0.f;
    #pragma unroll
    for (int q = 0; q < TOPK; ++q) {
      const float w = pbuf[row * TOPK + q];
      const float* mr = m + (size_t)ibuf[row * TOPK + q] * HDIM + seg * 16;
      const float4* mv = reinterpret_cast<const float4*>(mr);
      #pragma unroll
      for (int v4 = 0; v4 < 4; ++v4) {
        const float4 mm = mv[v4];
        oacc[v4 * 4 + 0] += w * mm.x;
        oacc[v4 * 4 + 1] += w * mm.y;
        oacc[v4 * 4 + 2] += w * mm.z;
        oacc[v4 * 4 + 3] += w * mm.w;
      }
    }
    __hip_bfloat16* fo = feat + (size_t)(grow0 + row) * FDIM + seg * 16;
    #pragma unroll
    for (int hh = 0; hh < 16; ++hh) fo[hh] = __float2bfloat16(oacc[hh]);
  }
}

// ---------------------------------------------------------------------------
// Final GEMM: out = feat[N,544]bf16 @ pred_w + pred_b + u_norm (f32 residual)
// One wave per 16x16 output tile; K = 544 = 17 WMMA steps.
// Grid: (N/16)*(H/16)/8 = 2048 blocks of 256 threads.
// ---------------------------------------------------------------------------
__global__ void __launch_bounds__(256) mtnet_pred(
    const __hip_bfloat16* __restrict__ feat,
    const __hip_bfloat16* __restrict__ wt,     // pred_w^T bf16 [H, F]
    const float* __restrict__ bias,
    const float* __restrict__ u_norm,
    float* __restrict__ out) {
  const int t    = threadIdx.x;
  const int wave = t >> 5;
  const int lane = t & 31;
  const int half = lane >> 4;
  const int l16  = lane & 15;
  const int gw = blockIdx.x * 8 + wave;
  const int tm = gw >> 4;                      // 0..1023 (N tiles)
  const int tn = gw & 15;                      // 0..15   (H tiles)

  const __hip_bfloat16* ap = feat + (size_t)(tm * 16 + l16) * FDIM + half * 8;
  const __hip_bfloat16* bp = wt   + (size_t)(tn * 16 + l16) * FDIM + half * 8;
  v8f acc = {};
  #pragma unroll
  for (int kk = 0; kk < 17; ++kk) {            // 17*32 = 544 = FDIM exactly
    v16bf a = load_bf16x16(ap + kk * 32);
    v16bf b = load_bf16x16(bp + kk * 32);
    acc = __builtin_amdgcn_wmma_f32_16x16x32_bf16(
        false, a, false, b, (short)0, acc, false, false);
  }
  const int col  = tn * 16 + l16;
  const float bc = bias[col];
  #pragma unroll
  for (int r = 0; r < 8; ++r) {
    const int grow = tm * 16 + r + 8 * half;
    out[(size_t)grow * HDIM + col] =
        acc[r] + bc + u_norm[(size_t)grow * HDIM + col];
  }
}

// ---------------------------------------------------------------------------
// Workspace layout (bytes, all 16B aligned):
//   [0)            u_norm f32  : N*H*4  = 16,777,216
//   [16777216)     feat  bf16  : N*F*2  = 17,825,792   ([o | u_norm | Q])
//   [34603008)     m     bf16  : M*H*2  =  2,097,152
//   [36700160)     wt    bf16  : H*F*2  =    278,528
// Total ~35.3 MB
// ---------------------------------------------------------------------------
extern "C" void kernel_launch(void* const* d_in, const int* in_sizes, int n_in,
                              void* d_out, int out_size, void* d_ws, size_t ws_size,
                              hipStream_t stream) {
  const float* Q  = (const float*)d_in[0];
  const float* u  = (const float*)d_in[1];
  const float* m  = (const float*)d_in[2];
  const float* pw = (const float*)d_in[3];
  const float* pb = (const float*)d_in[4];
  float* out = (float*)d_out;

  char* ws = (char*)d_ws;
  float*          u_norm = (float*)ws;
  __hip_bfloat16* feat   = (__hip_bfloat16*)(ws + 16777216u);
  __hip_bfloat16* mb     = (__hip_bfloat16*)(ws + 34603008u);
  __hip_bfloat16* wt     = (__hip_bfloat16*)(ws + 36700160u);

  mtnet_prep_norm<<<N_ROWS, 256, 0, stream>>>(u, Q, u_norm, feat);
  mtnet_prep_mbf<<<(MROWS * HDIM) / 256, 256, 0, stream>>>(m, mb);
  mtnet_prep_wt<<<FDIM, 256, 0, stream>>>(pw, wt);
  mtnet_score_topk_o<<<N_ROWS / 16, 256, K2_LDS, stream>>>(feat, mb, m);
  mtnet_pred<<<(N_ROWS / 16) * (HDIM / 16) / 8, 256, 0, stream>>>(
      feat, wt, pb, u_norm, out);
}
